// BigramLanguageModel_17231408791836
// MI455X (gfx1250) — matrix operand order repaired and checked by reading, not compiled, not run on hardware
//
#include <hip/hip_runtime.h>
#include <hip/hip_bf16.h>

typedef __attribute__((ext_vector_type(16))) _Float16 v16h;
typedef __attribute__((ext_vector_type(8)))  float    v8f;

#define BN 131072
#define TN 8
#define VN 65
#define CN 32
#define HN 4
#define DN 8
#define SCALEF 0.17677669529663687f   /* C^-0.5 (faithful to reference bug) */

#define ITERS 4
#define ROWS_PER_BLOCK (16 * ITERS)        /* 64  */
#define NBLOCKS (BN / ROWS_PER_BLOCK)      /* 2048 */
#define VP 80                               /* vocab padded to 5 N-tiles */

/* ---- A fragment: 16x32 f16, rows m=lane&15; lane half selects K subgroups ---- */
__device__ __forceinline__ v16h frag_a_from_lds(const float* t, int lane) {
  v16h a;
  const int m  = lane & 15;
  const int kb = (lane >> 4) * 8;
#pragma unroll
  for (int i = 0; i < 8; ++i) a[i]     = (_Float16)t[m * CN + kb + i];
#pragma unroll
  for (int i = 0; i < 8; ++i) a[8 + i] = (_Float16)t[m * CN + 16 + kb + i];
  return a;
}

/* ---- B fragment: 32x16 f16, col n=(lane&15)+n0; lane half selects K subgroups ---- */
__device__ __forceinline__ v16h frag_b_from_lds(const _Float16* W, int stride, int n0, int lane) {
  v16h b;
  const int n  = (lane & 15) + n0;
  const int kb = (lane >> 4) * 8;
#pragma unroll
  for (int i = 0; i < 16; ++i) b[i] = W[((i < 8) ? (kb + i) : (8 + kb + i)) * stride + n];
  return b;
}

/* ---- D fragment store: lane=(col), VGPR r = row r + 8*(lane>>4) ---- */
__device__ __forceinline__ void store_frag(float* t, int stride, int n0, v8f c, int lane) {
  const int n  = (lane & 15) + n0;
  const int mb = (lane >> 4) * 8;
#pragma unroll
  for (int r = 0; r < 8; ++r) t[(mb + r) * stride + n] = c[r];
}

__device__ __forceinline__ v8f wmma32(v16h a, v16h b, v8f c) {
  return __builtin_amdgcn_wmma_f32_16x16x32_f16(false, a, false, b, (short)0, c, false, false);
}

__global__ __launch_bounds__(256) void gpt_fwd_kernel(
    const int* __restrict__ idx, const int* __restrict__ targets,
    const float* __restrict__ tok_emb, const float* __restrict__ pos_emb,
    const float* __restrict__ wq, const float* __restrict__ wk, const float* __restrict__ wv,
    const float* __restrict__ wff, const float* __restrict__ bff,
    const float* __restrict__ wlm, const float* __restrict__ blm,
    float* __restrict__ logits_out, float* __restrict__ partial)
{
  __shared__ _Float16 s_wq[CN * CN], s_wk[CN * CN], s_wv[CN * CN], s_wff[CN * CN];
  __shared__ _Float16 s_wlm[CN * VP];
  __shared__ float s_tok[VN * CN];
  __shared__ float s_pos[TN * CN];
  __shared__ float s_bff[CN];
  __shared__ float s_blm[VP];
  __shared__ float s_arena[8][2048];   /* per-wave: x/o/h tile (512) + q/k/v (3*512); logits alias q.. */
  __shared__ float s_red[128];

  const int tid  = threadIdx.x;
  const int lane = tid & 31;
  const int wave = tid >> 5;

  /* ---------------- stage weights (once per block) ---------------- */
  for (int i = tid; i < CN * CN; i += 256) {
    const int h = i >> 8, c = (i >> 3) & 31, d = i & 7;
    const int dst = c * CN + h * DN + d;          /* W[c][h*8+d] = w[h][c][d] */
    s_wq[dst] = (_Float16)wq[i];
    s_wk[dst] = (_Float16)wk[i];
    s_wv[dst] = (_Float16)wv[i];
    s_wff[i]  = (_Float16)wff[i];                 /* [in][out] row-major already */
  }
  for (int i = tid; i < CN * VP; i += 256) {
    const int c = i / VP, n = i % VP;
    s_wlm[i] = (_Float16)((n < VN) ? wlm[c * VN + n] : 0.0f);
  }
  for (int i = tid; i < VN * CN; i += 256) s_tok[i] = tok_emb[i];
  if (tid < TN * CN) s_pos[tid] = pos_emb[tid];
  if (tid < CN)      s_bff[tid] = bff[tid];
  if (tid < VP)      s_blm[tid] = (tid < VN) ? blm[tid] : 0.0f;
  __syncthreads();

  /* ---------------- hoist constant B fragments into registers ---------------- */
  v16h bqkv[6];
  bqkv[0] = frag_b_from_lds(s_wq, CN, 0, lane);  bqkv[1] = frag_b_from_lds(s_wq, CN, 16, lane);
  bqkv[2] = frag_b_from_lds(s_wk, CN, 0, lane);  bqkv[3] = frag_b_from_lds(s_wk, CN, 16, lane);
  bqkv[4] = frag_b_from_lds(s_wv, CN, 0, lane);  bqkv[5] = frag_b_from_lds(s_wv, CN, 16, lane);
  v16h bffv[2];
  bffv[0] = frag_b_from_lds(s_wff, CN, 0, lane); bffv[1] = frag_b_from_lds(s_wff, CN, 16, lane);
  v16h blmv[5];
#pragma unroll
  for (int j = 0; j < 5; ++j) blmv[j] = frag_b_from_lds(s_wlm, VP, 16 * j, lane);

  float* xt = s_arena[wave];           /* x tile, reused as o tile then h tile */
  float* qt = s_arena[wave] + 512;
  float* kt = s_arena[wave] + 1024;
  float* vt = s_arena[wave] + 1536;
  float* lt = s_arena[wave] + 512;     /* 16x80 logits tile aliases q/k/v (free by then) */

  float nll_acc = 0.0f;
  const v8f zero = {};

  for (int it = 0; it < ITERS; ++it) {
    const long long r0 = (long long)blockIdx.x * ROWS_PER_BLOCK + it * 16 + wave * 2;

    /* ---- x = tok_emb[idx] + pos_emb : tile rows 0-7 -> batch r0, 8-15 -> r0+1 ---- */
#pragma unroll
    for (int j = 0; j < 16; ++j) {
      const long long b = r0 + (j >> 3);
      const int t = j & 7;
      const int tok = idx[b * TN + t];
      xt[j * CN + lane] = s_tok[tok * CN + lane] + s_pos[t * CN + lane];
    }
    __syncthreads();

    /* ---- fused QKV: 6 WMMAs ---- */
    {
      const v16h ax = frag_a_from_lds(xt, lane);
      store_frag(qt, CN, 0,  wmma32(ax, bqkv[0], zero), lane);
      store_frag(qt, CN, 16, wmma32(ax, bqkv[1], zero), lane);
      store_frag(kt, CN, 0,  wmma32(ax, bqkv[2], zero), lane);
      store_frag(kt, CN, 16, wmma32(ax, bqkv[3], zero), lane);
      store_frag(vt, CN, 0,  wmma32(ax, bqkv[4], zero), lane);
      store_frag(vt, CN, 16, wmma32(ax, bqkv[5], zero), lane);
    }
    __syncthreads();

    /* ---- causal attention, T=8 D=8: 64 (seq,head,t) rows over 32 lanes ---- */
#pragma unroll
    for (int p = 0; p < 2; ++p) {
      const int item = lane + 32 * p;
      const int seq  = item >> 5;
      const int head = (item >> 3) & 3;
      const int t    = item & 7;
      const float* qrow = qt + (seq * 8 + t) * CN + head * DN;
      float att[TN];
      float mx = -3.0e38f;
#pragma unroll
      for (int s = 0; s < TN; ++s) {
        const float* krow = kt + (seq * 8 + s) * CN + head * DN;
        float acc = 0.0f;
#pragma unroll
        for (int d = 0; d < DN; ++d) acc += qrow[d] * krow[d];
        acc = (s <= t) ? acc * SCALEF : -3.0e38f;
        att[s] = acc;
        mx = fmaxf(mx, acc);
      }
      float sum = 0.0f;
#pragma unroll
      for (int s = 0; s < TN; ++s) {
        const float e = (s <= t) ? __expf(att[s] - mx) : 0.0f;
        att[s] = e;
        sum += e;
      }
      const float inv = 1.0f / sum;
      float o[DN] = {0, 0, 0, 0, 0, 0, 0, 0};
#pragma unroll
      for (int s = 0; s < TN; ++s) {
        const float w = att[s] * inv;
        const float* vrow = vt + (seq * 8 + s) * CN + head * DN;
#pragma unroll
        for (int d = 0; d < DN; ++d) o[d] += w * vrow[d];
      }
      float* orow = xt + (seq * 8 + t) * CN + head * DN;  /* head-concat layout */
#pragma unroll
      for (int d = 0; d < DN; ++d) orow[d] = o[d];
    }
    __syncthreads();

    /* ---- feed-forward: 2 WMMAs + bias + ReLU ---- */
    {
      const v16h ao = frag_a_from_lds(xt, lane);
      const v8f hlo = wmma32(ao, bffv[0], zero);
      const v8f hhi = wmma32(ao, bffv[1], zero);
      const int n  = lane & 15;
      const int mb = (lane >> 4) * 8;
#pragma unroll
      for (int r = 0; r < 8; ++r) {
        xt[(mb + r) * CN + n]      = fmaxf(hlo[r] + s_bff[n], 0.0f);
        xt[(mb + r) * CN + n + 16] = fmaxf(hhi[r] + s_bff[n + 16], 0.0f);
      }
    }
    __syncthreads();

    /* ---- LM head: 5 WMMAs (N padded to 80), bias, write logits ---- */
    {
      const v16h ah = frag_a_from_lds(xt, lane);
      v8f lf[5];
#pragma unroll
      for (int j = 0; j < 5; ++j) lf[j] = wmma32(ah, blmv[j], zero);

      const int ncol = lane & 15;
      const int hi   = lane >> 4;
      const long long bb = r0 + hi;       /* this lane's batch row for its 8 M-rows */
#pragma unroll
      for (int j = 0; j < 5; ++j) {
        const int n = ncol + 16 * j;
        if (n < VN) {
          const float bias = s_blm[n];
#pragma unroll
          for (int r = 0; r < 8; ++r) {
            const float val = lf[j][r] + bias;
            lt[(8 * hi + r) * VP + n] = val;
            logits_out[(size_t)(bb * TN + r) * VN + n] = val;
          }
        }
      }
    }
    __syncthreads();

    /* ---- cross-entropy: lanes 0-15, one tile row each ---- */
    if (lane < 16) {
      const int m = lane;
      const long long b = r0 + (m >> 3);
      const int t = m & 7;
      const int tgt = targets[b * TN + t];
      const float* row = lt + m * VP;
      float mx = row[0];
      for (int n = 1; n < VN; ++n) mx = fmaxf(mx, row[n]);
      float sum = 0.0f;
      for (int n = 0; n < VN; ++n) sum += __expf(row[n] - mx);
      nll_acc += (mx + __logf(sum)) - row[tgt];
    }
    __syncthreads();
  }

  /* ---- deterministic fixed-order block reduction ---- */
  if (lane < 16) s_red[wave * 16 + lane] = nll_acc;
  __syncthreads();
  if (tid == 0) {
    float s = 0.0f;
    for (int i = 0; i < 128; ++i) s += s_red[i];
    partial[blockIdx.x] = s;
  }
}

__global__ __launch_bounds__(256) void reduce_loss_kernel(const float* __restrict__ partial,
                                                          float* __restrict__ out_loss)
{
  __shared__ float s[256];
  const int tid = threadIdx.x;
  const int chunk = NBLOCKS / 256;   /* 8 */
  float a = 0.0f;
  for (int i = 0; i < chunk; ++i) a += partial[tid * chunk + i];
  s[tid] = a;
  __syncthreads();
  if (tid == 0) {
    float tot = 0.0f;
    for (int i = 0; i < 256; ++i) tot += s[i];
    *out_loss = tot * (1.0f / (float)(BN * TN));
  }
}

extern "C" void kernel_launch(void* const* d_in, const int* in_sizes, int n_in,
                              void* d_out, int out_size, void* d_ws, size_t ws_size,
                              hipStream_t stream) {
  const int*   idx  = (const int*)d_in[0];
  const int*   tgt  = (const int*)d_in[1];
  const float* tok  = (const float*)d_in[2];
  const float* pos  = (const float*)d_in[3];
  const float* wq   = (const float*)d_in[4];
  const float* wk   = (const float*)d_in[5];
  const float* wv   = (const float*)d_in[6];
  const float* wff  = (const float*)d_in[7];
  const float* bff  = (const float*)d_in[8];
  const float* wlm  = (const float*)d_in[9];
  const float* blm  = (const float*)d_in[10];

  float* logits  = (float*)d_out;
  float* loss    = logits + (size_t)BN * TN * VN;
  float* partial = (float*)d_ws;

  gpt_fwd_kernel<<<dim3(NBLOCKS), dim3(256), 0, stream>>>(
      idx, tgt, tok, pos, wq, wk, wv, wff, bff, wlm, blm, logits, partial);
  reduce_loss_kernel<<<dim3(1), dim3(256), 0, stream>>>(partial, loss);
}
